// MarkovConvolution_27410481283753
// MI455X (gfx1250) — matrix operand order repaired and checked by reading, not compiled, not run on hardware
//
#include <hip/hip_runtime.h>
#include <hip/hip_fp16.h>
#include <hip/hip_bf16.h>

// CDNA5 / gfx1250, wave32
typedef __attribute__((ext_vector_type(16))) _Float16 v16h;
typedef __attribute__((ext_vector_type(8)))  float    v8f;
typedef __attribute__((ext_vector_type(4)))  float    v4f;
typedef __attribute__((ext_vector_type(4)))  int      v4i;

#define BB      2048
#define LL      64
#define NN      1024
#define FF      8
#define SE      16
#define EE      32
#define SS      20
#define NSH     56      // 32 + 8 + 16
#define NCW     59      // conv kernel width
#define KPAD    64      // padded K for WMMA (2 x K=32 steps)
#define ROWS    16      // b-rows per block (WMMA M)

// ======================= split path (kernel A): logits -> ws =======================
#define THREADS_A 256                         // 8 waves
#define NWAVES_A  (THREADS_A / 32)
#define COLS_A    256                         // n-columns per block
#define TILES_A   ((COLS_A / 16) / NWAVES_A)  // 2 n-tiles per wave
#define FPITCH    (COLS_A + 8)                // LDS fout row pitch (floats)
#define CHUNKS_A  ((ROWS * (COLS_A / 4)) / THREADS_A)   // 4 float4 chunks / thread
// LDS: fout 16*264*4 = 16896 B ; shA 16*64*2 = 2048 B ; wl 256 B  => ~19.2 KB
#define SMEM_FOUT_BYTES (ROWS * FPITCH * 4)
#define SMEM_SHA_BYTES  (ROWS * KPAD * 2)
#define SMEM_A_TOTAL    (SMEM_FOUT_BYTES + SMEM_SHA_BYTES + LL * 4)

__global__ void logits_kernel(
    const float* __restrict__ x,          // [B, L, N]
    const float* __restrict__ x_dist,     // [N]
    const float* __restrict__ x_features, // [B, F]
    const float* __restrict__ x_markov,   // [B, N]
    const int*   __restrict__ stops,      // [B, S]
    const int*   __restrict__ x_week,     // [B]
    const float* __restrict__ stop_emb,   // [N, SE]
    const float* __restrict__ week_emb,   // [NW, E]
    const float* __restrict__ conv_w,     // [N, 59]
    const float* __restrict__ conv_b,     // [N]
    const float* __restrict__ fc1_w,      // [L]
    const float* __restrict__ fc1_b,      // scalar
    float* __restrict__ ws_logits)        // [B, N]
{
    __shared__ char smem[SMEM_A_TOTAL];
    float*    fout = (float*)smem;
    _Float16* shA  = (_Float16*)(smem + SMEM_FOUT_BYTES);
    float*    wl   = (float*)(smem + SMEM_FOUT_BYTES + SMEM_SHA_BYTES);

    const int tid    = threadIdx.x;
    const int n_base = blockIdx.x * COLS_A;
    const int b0     = blockIdx.y * ROWS;
    const int wave   = tid >> 5;
    const int lane   = tid & 31;
    const int laneN  = lane & 15;
    const int hi     = lane >> 4;

    // ---- Phase 0: stage fc1_w and the padded f16 "shared" A tile ----
    if (tid < LL) wl[tid] = fc1_w[tid];
    for (int i = tid; i < ROWS * KPAD; i += THREADS_A) {
        const int r = i >> 6;
        const int k = i & (KPAD - 1);
        const int b = b0 + r;
        float v;
        if (k < EE) {
            v = week_emb[x_week[b] * EE + k];
        } else if (k < EE + FF) {
            v = x_features[b * FF + (k - EE)];
        } else if (k < NSH) {
            const int c = k - (EE + FF);
            float s = 0.f;
            #pragma unroll
            for (int j = 0; j < SS; ++j)
                s += stop_emb[stops[b * SS + j] * SE + c];
            v = s;
        } else {
            v = 0.f;
        }
        shA[r * KPAD + k] = (_Float16)v;
    }
    __syncthreads();

    const float fb = fc1_b[0];

    // ---- Phase 1a: fc1 stream of x (512 MB device-wide) -> fout in LDS ----
    // float4 NT loads, consecutive tids = consecutive 16B chunks => 512B coalesced
    // transactions; every byte of x read exactly once; L2 left to conv_w.
    {
        v4f acc[CHUNKS_A];
        const v4f* xp[CHUNKS_A];
        int dst[CHUNKS_A];
        #pragma unroll
        for (int c = 0; c < CHUNKS_A; ++c) {
            const int g   = c * THREADS_A + tid;        // 0..1023
            const int row = g >> 6;                     // 0..15
            const int ch  = g & (COLS_A / 4 - 1);       // 0..63
            xp[c]  = (const v4f*)(x + (long)(b0 + row) * LL * NN + n_base) + ch;
            dst[c] = row * FPITCH + ch * 4;
            acc[c] = (v4f){fb, fb, fb, fb};
        }
        for (int l = 0; l < LL; ++l) {
            const float w = wl[l];                      // uniform LDS broadcast
            const long off = (long)l * (NN / 4);
            #pragma unroll
            for (int c = 0; c < CHUNKS_A; ++c) {
                const v4f v = __builtin_nontemporal_load(xp[c] + off);
                acc[c] += v * w;
            }
        }
        #pragma unroll
        for (int c = 0; c < CHUNKS_A; ++c)
            *(v4f*)(fout + dst[c]) = acc[c];
    }
    __syncthreads();

    // ---- Phase 1b: WMMA GEMM (shared @ conv_w[:, :56]^T) + fused combine -> ws ----
    // A 16x32 f16 as loaded here (contiguous k, same order as B): lanes 0-15 K=0..15; lanes 16-31 K=16..31; M=lane&15.
    v16h a0, a1;
    {
        const _Float16* arow = shA + laneN * KPAD;
        const int kb = hi ? 16 : 0;
        #pragma unroll
        for (int j = 0; j < 8; ++j) {
            a0[j]     = arow[kb + j];
            a0[8 + j] = arow[kb + 8 + j];
            a1[j]     = arow[32 + kb + j];
            a1[8 + j] = arow[32 + kb + 8 + j];
        }
    }

    #pragma unroll
    for (int t = 0; t < TILES_A; ++t) {
        const int n_loc = (wave * TILES_A + t) * 16 + laneN;
        const int n     = n_base + n_loc;
        const float* cwrow = conv_w + n * NCW;

        // B 32x16 as loaded here (contiguous k, same order as A): lanes 0-15 K=0..15 of column n=lane; lanes 16-31 K=16..31.
        v16h bf0, bf1;
        {
            const int ks = hi ? 16 : 0;
            #pragma unroll
            for (int j = 0; j < 16; ++j) {
                const int k0 = ks + j;
                const int k1 = 32 + ks + j;
                bf0[j] = (_Float16)((k0 < NSH) ? cwrow[k0] : 0.f);
                bf1[j] = (_Float16)((k1 < NSH) ? cwrow[k1] : 0.f);
            }
        }

        v8f c = {};
        c = __builtin_amdgcn_wmma_f32_16x16x32_f16(false, a0, false, bf0, (short)0, c, false, false);
        c = __builtin_amdgcn_wmma_f32_16x16x32_f16(false, a1, false, bf1, (short)0, c, false, false);

        const float cw_out  = cwrow[NSH];
        const float cw_dist = cwrow[NSH + 1] * x_dist[n];
        const float cw_mk   = cwrow[NSH + 2];
        const float cb      = conv_b[n];

        // C/D layout: VGPR r -> M = r + 8*hi; N = lane&15.
        #pragma unroll
        for (int r = 0; r < 8; ++r) {
            const int m = r + 8 * hi;
            const int b = b0 + m;
            const float fo = fout[m * FPITCH + n_loc];
            const float logit = c[r]
                              + fo * cw_out
                              + cw_dist
                              + x_markov[(long)b * NN + n] * cw_mk
                              + cb;
            ws_logits[(long)b * NN + n] = logit;
        }
    }
}

// ======================= split path (kernel B): row log_softmax + mask =======================
#define THREADS_B 256    // 8 waves; each thread owns 4 columns (float4)

__global__ void softmax_kernel(const float* __restrict__ ws_logits,
                               const int*   __restrict__ x_mask,
                               float* __restrict__ out)
{
    __shared__ float red[THREADS_B / 32];   // 8
    const int b   = blockIdx.x;
    const int tid = threadIdx.x;
    const int wave = tid >> 5;
    const int lane = tid & 31;

    const v4f v = *((const v4f*)(ws_logits + (long)b * NN) + tid);

    float mx = fmaxf(fmaxf(v[0], v[1]), fmaxf(v[2], v[3]));
    #pragma unroll
    for (int off = 16; off >= 1; off >>= 1)
        mx = fmaxf(mx, __shfl_xor(mx, off, 32));
    if (lane == 0) red[wave] = mx;
    __syncthreads();
    float mall = red[0];
    #pragma unroll
    for (int j = 1; j < THREADS_B / 32; ++j) mall = fmaxf(mall, red[j]);
    __syncthreads();   // all reads of red done before reuse

    float s = __expf(v[0] - mall) + __expf(v[1] - mall)
            + __expf(v[2] - mall) + __expf(v[3] - mall);
    #pragma unroll
    for (int off = 16; off >= 1; off >>= 1)
        s += __shfl_xor(s, off, 32);
    if (lane == 0) red[wave] = s;
    __syncthreads();
    float sall = red[0];
    #pragma unroll
    for (int j = 1; j < THREADS_B / 32; ++j) sall += red[j];

    const float lse = mall + __logf(sall);

    const v4i msk = *((const v4i*)(x_mask + (long)b * NN) + tid);
    v4f lp;
    #pragma unroll
    for (int i = 0; i < 4; ++i)
        lp[i] = msk[i] ? -1.0e8f : (v[i] - lse);
    __builtin_nontemporal_store(lp, (v4f*)(out + (long)b * NN) + tid);
}

// ======================= fallback: fully fused single kernel (no ws needed) =======================
#define THREADS 512
#define NWAVES  (THREADS / 32)
#define TILES_PER_WAVE ((NN / 16) / NWAVES)
#define LPITCH  (NN + 8)
#define CHUNKS_PER_THREAD ((ROWS * (NN / 4)) / THREADS)
#define SMEM_LOGITS_BYTES (ROWS * LPITCH * 4)
#define SMEM_TOTAL        (SMEM_LOGITS_BYTES + SMEM_SHA_BYTES + LL * 4)

__global__ void markov_conv_fused_kernel(
    const float* __restrict__ x, const float* __restrict__ x_dist,
    const float* __restrict__ x_features, const float* __restrict__ x_markov,
    const int* __restrict__ stops, const int* __restrict__ x_week,
    const int* __restrict__ x_mask, const float* __restrict__ stop_emb,
    const float* __restrict__ week_emb, const float* __restrict__ conv_w,
    const float* __restrict__ conv_b, const float* __restrict__ fc1_w,
    const float* __restrict__ fc1_b, float* __restrict__ out)
{
    extern __shared__ char smem[];
    float*    logits = (float*)smem;
    _Float16* shA    = (_Float16*)(smem + SMEM_LOGITS_BYTES);
    float*    wl     = (float*)(smem + SMEM_LOGITS_BYTES + SMEM_SHA_BYTES);

    const int tid  = threadIdx.x;
    const int b0   = blockIdx.x * ROWS;
    const int wave = tid >> 5;
    const int lane = tid & 31;
    const int laneN = lane & 15;
    const int hi    = lane >> 4;

    if (tid < LL) wl[tid] = fc1_w[tid];
    for (int i = tid; i < ROWS * KPAD; i += THREADS) {
        const int r = i >> 6;
        const int k = i & (KPAD - 1);
        const int b = b0 + r;
        float v;
        if (k < EE) v = week_emb[x_week[b] * EE + k];
        else if (k < EE + FF) v = x_features[b * FF + (k - EE)];
        else if (k < NSH) {
            const int c = k - (EE + FF);
            float s = 0.f;
            #pragma unroll
            for (int j = 0; j < SS; ++j) s += stop_emb[stops[b * SS + j] * SE + c];
            v = s;
        } else v = 0.f;
        shA[r * KPAD + k] = (_Float16)v;
    }
    __syncthreads();

    const float fb = fc1_b[0];
    {
        v4f acc[CHUNKS_PER_THREAD];
        const v4f* xp[CHUNKS_PER_THREAD];
        int dst[CHUNKS_PER_THREAD];
        #pragma unroll
        for (int c = 0; c < CHUNKS_PER_THREAD; ++c) {
            const int g   = c * THREADS + tid;
            const int row = g >> 8;
            const int ch  = g & (NN / 4 - 1);
            xp[c]  = (const v4f*)(x + (long)(b0 + row) * LL * NN) + ch;
            dst[c] = row * LPITCH + ch * 4;
            acc[c] = (v4f){fb, fb, fb, fb};
        }
        for (int l = 0; l < LL; ++l) {
            const float w = wl[l];
            const long off = (long)l * (NN / 4);
            #pragma unroll
            for (int c = 0; c < CHUNKS_PER_THREAD; ++c)
                acc[c] += __builtin_nontemporal_load(xp[c] + off) * w;
        }
        #pragma unroll
        for (int c = 0; c < CHUNKS_PER_THREAD; ++c)
            *(v4f*)(logits + dst[c]) = acc[c];
    }
    __syncthreads();

    v16h a0, a1;
    {
        const _Float16* arow = shA + laneN * KPAD;
        const int kb = hi ? 16 : 0;
        #pragma unroll
        for (int j = 0; j < 8; ++j) {
            a0[j]     = arow[kb + j];
            a0[8 + j] = arow[kb + 8 + j];
            a1[j]     = arow[32 + kb + j];
            a1[8 + j] = arow[32 + kb + 8 + j];
        }
    }
    #pragma unroll
    for (int t = 0; t < TILES_PER_WAVE; ++t) {
        const int n = (wave * TILES_PER_WAVE + t) * 16 + laneN;
        const float* cwrow = conv_w + n * NCW;
        v16h bf0, bf1;
        {
            const int ks = hi ? 16 : 0;
            #pragma unroll
            for (int j = 0; j < 16; ++j) {
                const int k0 = ks + j, k1 = 32 + ks + j;
                bf0[j] = (_Float16)((k0 < NSH) ? cwrow[k0] : 0.f);
                bf1[j] = (_Float16)((k1 < NSH) ? cwrow[k1] : 0.f);
            }
        }
        v8f c = {};
        c = __builtin_amdgcn_wmma_f32_16x16x32_f16(false, a0, false, bf0, (short)0, c, false, false);
        c = __builtin_amdgcn_wmma_f32_16x16x32_f16(false, a1, false, bf1, (short)0, c, false, false);
        const float cw_out  = cwrow[NSH];
        const float cw_dist = cwrow[NSH + 1] * x_dist[n];
        const float cw_mk   = cwrow[NSH + 2];
        const float cb      = conv_b[n];
        #pragma unroll
        for (int r = 0; r < 8; ++r) {
            const int m = r + 8 * hi;
            const int b = b0 + m;
            const float fo = logits[m * LPITCH + n];
            logits[m * LPITCH + n] = c[r] + fo * cw_out + cw_dist
                                   + x_markov[(long)b * NN + n] * cw_mk + cb;
        }
    }
    __syncthreads();

    {
        const int r = wave;
        const int b = b0 + r;
        const float* lrow = logits + r * LPITCH;
        float mx = -3.402823466e+38f;
        for (int nc = lane; nc < NN; nc += 32) mx = fmaxf(mx, lrow[nc]);
        #pragma unroll
        for (int off = 16; off >= 1; off >>= 1) mx = fmaxf(mx, __shfl_xor(mx, off, 32));
        float s = 0.f;
        for (int nc = lane; nc < NN; nc += 32) s += __expf(lrow[nc] - mx);
        #pragma unroll
        for (int off = 16; off >= 1; off >>= 1) s += __shfl_xor(s, off, 32);
        const float lse = mx + __logf(s);
        for (int nc = lane; nc < NN; nc += 32) {
            const float lp = lrow[nc] - lse;
            const float res = x_mask[(long)b * NN + nc] ? -1.0e8f : lp;
            __builtin_nontemporal_store(res, out + (long)b * NN + nc);
        }
    }
}

extern "C" void kernel_launch(void* const* d_in, const int* in_sizes, int n_in,
                              void* d_out, int out_size, void* d_ws, size_t ws_size,
                              hipStream_t stream) {
    (void)in_sizes; (void)n_in; (void)out_size;
    const float* x          = (const float*)d_in[0];
    const float* x_dist     = (const float*)d_in[1];
    const float* x_features = (const float*)d_in[2];
    const float* x_markov   = (const float*)d_in[3];
    const int*   stops      = (const int*)  d_in[4];
    const int*   x_week     = (const int*)  d_in[5];
    const int*   x_mask     = (const int*)  d_in[6];
    const float* stop_emb   = (const float*)d_in[7];
    const float* week_emb   = (const float*)d_in[8];
    const float* conv_w     = (const float*)d_in[9];
    const float* conv_b     = (const float*)d_in[10];
    const float* fc1_w      = (const float*)d_in[11];
    const float* fc1_b      = (const float*)d_in[12];
    float* out = (float*)d_out;

    const size_t need = (size_t)BB * NN * sizeof(float);   // 8 MB
    if (ws_size >= need) {
        float* ws_logits = (float*)d_ws;
        dim3 gridA(NN / COLS_A, BB / ROWS);   // (4, 128) = 512 blocks
        logits_kernel<<<gridA, THREADS_A, 0, stream>>>(
            x, x_dist, x_features, x_markov, stops, x_week,
            stop_emb, week_emb, conv_w, conv_b, fc1_w, fc1_b, ws_logits);
        softmax_kernel<<<dim3(BB), THREADS_B, 0, stream>>>(ws_logits, x_mask, out);
    } else {
        dim3 grid(BB / ROWS);
        markov_conv_fused_kernel<<<grid, THREADS, SMEM_TOTAL, stream>>>(
            x, x_dist, x_features, x_markov, stops, x_week, x_mask,
            stop_emb, week_emb, conv_w, conv_b, fc1_w, fc1_b, out);
    }
}
